// HypergraphAttentionIsomorphism_850403524773
// MI455X (gfx1250) — compile-verified, matrix operand order checked by reading, not run on hardware
//
#include <hip/hip_runtime.h>
#include <hip/hip_bf16.h>
#include <math.h>

// ---------------------------------------------------------------------------
// HypergraphAttention pipeline for MI455X (gfx1250, wave32, WMMA).
//
// Roofline: two 12000x12000x256 GEMMs dominate (149 GFLOP total). The two
// fp32 NxN matrices (2 x 576 MB) must stream from HBM once -> ~50us floor at
// 23.3 TB/s. Intensity vs HBM ~128 FLOP/B -> only bf16 WMMA (f32 accum) can
// keep up; fp32 VALU would be ~15x too slow. NxN operands are converted
// f32->bf16 in registers inside the GEMM (no extra HBM pass).
// ---------------------------------------------------------------------------

typedef __attribute__((ext_vector_type(16))) __bf16 v16bf;
typedef __attribute__((ext_vector_type(8)))  __bf16 v8bf;
typedef __attribute__((ext_vector_type(8)))  float  v8f;

#define N_NODES 12000
#define F_DIM   256

// ---------------------------------------------------------------------------
// scores[i] = dot(input[i,:], attn)   (one wave per row, shuffle reduce)
// ---------------------------------------------------------------------------
__global__ void scores_kernel(const float* __restrict__ input,
                              const float* __restrict__ attn,
                              float* __restrict__ scores, int N, int F) {
  int row = blockIdx.x;
  int lane = threadIdx.x;  // blockDim.x == 32
  float s = 0.f;
  const float* rp = input + (size_t)row * F;
  for (int j = lane; j < F; j += 32) s += rp[j] * attn[j];
  for (int off = 16; off > 0; off >>= 1) s += __shfl_down(s, off, 32);
  if (lane == 0) scores[row] = s;
}

// ---------------------------------------------------------------------------
// Global softmax reduction over scores[0..N): red[0]=max, red[1]=1/sum(exp)
// ---------------------------------------------------------------------------
__global__ void softmax_reduce_kernel(const float* __restrict__ scores,
                                      float* __restrict__ red, int N) {
  __shared__ float sh[1024];
  int t = threadIdx.x;
  float m = -INFINITY;
  for (int i = t; i < N; i += 1024) m = fmaxf(m, scores[i]);
  sh[t] = m; __syncthreads();
  for (int s = 512; s > 0; s >>= 1) {
    if (t < s) sh[t] = fmaxf(sh[t], sh[t + s]);
    __syncthreads();
  }
  float mx = sh[0]; __syncthreads();
  float sum = 0.f;
  for (int i = t; i < N; i += 1024) sum += expf(scores[i] - mx);
  sh[t] = sum; __syncthreads();
  for (int s = 512; s > 0; s >>= 1) {
    if (t < s) sh[t] += sh[t + s];
    __syncthreads();
  }
  if (t == 0) { red[0] = mx; red[1] = 1.0f / sh[0]; }
}

// ---------------------------------------------------------------------------
// dst[c][r] = bf16( src[r][c] * (scores? softmax(scores)[r] : 1) )
// Produces K-major bf16 "B^T" operands so a lane's WMMA B-fragment is two
// contiguous 16B loads. LDS-tiled so both global sides are coalesced.
// ---------------------------------------------------------------------------
__global__ void transpose_scale_bf16(const float* __restrict__ src,  // [R][C]
                                     __bf16* __restrict__ dst,       // [C][R]
                                     const float* __restrict__ scores,
                                     const float* __restrict__ red,
                                     int R, int C) {
  __shared__ float tile[32][33];
  int rb = blockIdx.x * 32, cb = blockIdx.y * 32;
  int tx = threadIdx.x, ty = threadIdx.y;  // (32, 8)
  for (int i = ty; i < 32; i += 8) {
    int r = rb + i, c = cb + tx;
    tile[i][tx] = (r < R && c < C) ? src[(size_t)r * C + c] : 0.f;
  }
  __syncthreads();
  for (int i = ty; i < 32; i += 8) {
    int c = cb + i, r = rb + tx;
    if (c < C && r < R) {
      float v = tile[tx][i];
      if (scores) v *= expf(scores[r] - red[0]) * red[1];
      dst[(size_t)c * R + r] = (__bf16)v;
    }
  }
}

// ---------------------------------------------------------------------------
// C[M,256] = A_f32[M,K] @ B (given as Bt bf16 [256][K], Bt[n][k]=B[k][n])
//
// LDS-free register GEMM. Block = 256 threads = 8 waves; wave(wr,wc) owns a
// 32x64 C tile = 2x4 WMMA 16x16 tiles (64 accum VGPRs/lane). Per K-step of
// 32: A rows loaded as fp32 (the two half-lanes of a row cover one full 128B
// line), converted to bf16 fragments per the 16-bit A VGPR layout
// (lane half h, elem e: K = 8h+e | 16+8h+(e-8)); B fragments are two 16B
// loads from the K-major bf16 operand (L2-resident). 8 WMMAs per step.
// All 8 waves of a block share the same 64 A rows -> A streams from HBM once.
//
// Fragment assembly uses __builtin_convertvector (packed v_cvt_pk_bf16_f32)
// and __builtin_shufflevector (register-coalesced halves, no v_mov_b16).
//
// EPI 0: store bf16 transposed (outT[n][m]) -> feeds next GEMM's B operand.
// EPI 1: store f32 row-major of (acc + alpha*input).
// EPI 2: store f32 row-major.
// ---------------------------------------------------------------------------
template <int EPI>
__global__ __launch_bounds__(256) void gemm_wmma_bf16(
    const float* __restrict__ A,     // [M][K] f32 row-major
    const __bf16* __restrict__ Bt,   // [256][K] bf16
    float* __restrict__ outF,        // EPI 1,2
    __bf16* __restrict__ outT,       // EPI 0, [256][M]
    const float* __restrict__ inp,   // EPI 1, [M][256]
    const float* __restrict__ alphaPtr,
    int M, int K) {
  const int lane = threadIdx.x & 31;
  const int wave = threadIdx.x >> 5;
  const int wr = wave >> 2;       // 0..1 : which 32-row group
  const int wc = wave & 3;        // 0..3 : which 64-col group
  const int half = lane >> 4;     // 0..1
  const int lm = lane & 15;       // 0..15
  const int row0 = blockIdx.x * 64 + wr * 32;
  if (row0 >= M) return;          // wave-uniform exit (EXEC stays all-ones)
  const int col0 = wc * 64;

  float alpha = 0.f;
  if (EPI == 1) alpha = alphaPtr[0];

  v8f acc[2][4];
#pragma unroll
  for (int mi = 0; mi < 2; ++mi)
#pragma unroll
    for (int ni = 0; ni < 4; ++ni) acc[mi][ni] = (v8f)(0.f);

  const int k_lo = 8 * half;        // elems 0..7
  const int k_hi = 16 + 8 * half;   // elems 8..15

  const float* arow[2];
  arow[0] = A + (size_t)(row0 + lm) * K;
  arow[1] = A + (size_t)(row0 + 16 + lm) * K;
  const __bf16* brow[4];
#pragma unroll
  for (int ni = 0; ni < 4; ++ni)
    brow[ni] = Bt + (size_t)(col0 + ni * 16 + lm) * K;

  for (int k0 = 0; k0 < K; k0 += 32) {
    // Prefetch the streaming fp32 matrix ~1KB ahead (global_prefetch_b8;
    // speculative TH -> OOB prefetches silently dropped).
    __builtin_prefetch(arow[0] + k0 + 256, 0, 0);
    __builtin_prefetch(arow[1] + k0 + 256, 0, 0);

    v16bf afrag[2];
#pragma unroll
    for (int mi = 0; mi < 2; ++mi) {
      const float* p = arow[mi] + k0;
      v8f x0 = *(const v8f*)(p + k_lo);   // 32B aligned (K % 8 == 0 rows)
      v8f x1 = *(const v8f*)(p + k_hi);
      v8bf lo = __builtin_convertvector(x0, v8bf);  // packed cvt_pk_bf16_f32
      v8bf hi = __builtin_convertvector(x1, v8bf);
      afrag[mi] = __builtin_shufflevector(lo, hi, 0, 1, 2, 3, 4, 5, 6, 7,
                                          8, 9, 10, 11, 12, 13, 14, 15);
    }
    v16bf bfrag[4];
#pragma unroll
    for (int ni = 0; ni < 4; ++ni) {
      const __bf16* p = brow[ni] + k0;
      v8bf y0 = *(const v8bf*)(p + k_lo);  // 16B loads
      v8bf y1 = *(const v8bf*)(p + k_hi);
      bfrag[ni] = __builtin_shufflevector(y0, y1, 0, 1, 2, 3, 4, 5, 6, 7,
                                          8, 9, 10, 11, 12, 13, 14, 15);
    }
#pragma unroll
    for (int mi = 0; mi < 2; ++mi)
#pragma unroll
      for (int ni = 0; ni < 4; ++ni)
        acc[mi][ni] = __builtin_amdgcn_wmma_f32_16x16x32_bf16(
            false, afrag[mi], false, bfrag[ni], (short)0, acc[mi][ni],
            false, false);
  }

  // Epilogue. C/D layout: VGPR v of lane (half,lm) holds row (8*half+v),
  // col lm within each 16x16 tile.
#pragma unroll
  for (int mi = 0; mi < 2; ++mi) {
#pragma unroll
    for (int ni = 0; ni < 4; ++ni) {
      const int col = col0 + ni * 16 + lm;
      if (EPI == 0) {
        const int rowb = row0 + mi * 16 + 8 * half;  // 8 contiguous rows
        v8bf v = __builtin_convertvector(acc[mi][ni], v8bf);
        *(v8bf*)(outT + (size_t)col * M + rowb) = v;  // one 16B store/lane
      } else {
#pragma unroll
        for (int e = 0; e < 8; ++e) {
          const int row = row0 + mi * 16 + 8 * half + e;
          float val = acc[mi][ni][e];
          if (EPI == 1) val += alpha * inp[(size_t)row * F_DIM + col];
          outF[(size_t)row * F_DIM + col] = val;
        }
      }
    }
  }
}

// ---------------------------------------------------------------------------
// Host-side orchestration. Workspace layout (bytes), total ~24.8 MB:
//   [0)         scores f32 [12000]
//   [48128)     red f32 [2]  (max, 1/sum)
//   [49152)     scaledT bf16 [256][12000]   = (softmax .* input)^T
//   [6193152)   Wt bf16 [256][256]          = weight^T
//   [6324224)   support4 f32 [12000][256]
//   [18612224)  supportT bf16 [256][12000]
// ---------------------------------------------------------------------------
extern "C" void kernel_launch(void* const* d_in, const int* in_sizes, int n_in,
                              void* d_out, int out_size, void* d_ws,
                              size_t ws_size, hipStream_t stream) {
  (void)in_sizes; (void)n_in; (void)out_size; (void)ws_size;
  const float* input     = (const float*)d_in[0];  // [12000,256]
  const float* adj       = (const float*)d_in[1];  // [12000,12000]
  const float* incidence = (const float*)d_in[2];  // [12000,12000]
  const float* weight    = (const float*)d_in[3];  // [256,256]
  const float* attn      = (const float*)d_in[4];  // [256,1]
  const float* alpha     = (const float*)d_in[5];  // [1]
  float* out = (float*)d_out;                      // [12000,256]

  const int N = N_NODES, F = F_DIM;
  char* ws = (char*)d_ws;
  float*  scores   = (float*)(ws + 0);
  float*  red      = (float*)(ws + 48128);
  __bf16* scaledT  = (__bf16*)(ws + 49152);
  __bf16* Wt       = (__bf16*)(ws + 6193152);
  float*  support4 = (float*)(ws + 6324224);
  __bf16* supportT = (__bf16*)(ws + 18612224);

  // 1) per-row attention scores
  scores_kernel<<<N, 32, 0, stream>>>(input, attn, scores, N, F);
  // 2) global softmax normalization constants
  softmax_reduce_kernel<<<1, 1024, 0, stream>>>(scores, red, N);
  // 3) scaledT = (softmax(scores) .* input)^T in bf16 (K-major B operand)
  transpose_scale_bf16<<<dim3((N + 31) / 32, (F + 31) / 32), dim3(32, 8), 0,
                         stream>>>(input, scaledT, scores, red, N, F);
  //    Wt = weight^T in bf16
  transpose_scale_bf16<<<dim3((F + 31) / 32, (F + 31) / 32), dim3(32, 8), 0,
                         stream>>>(weight, Wt, nullptr, nullptr, F, F);

  const int gemm_grid = (N + 63) / 64;
  // 4) support4 = adj @ scaled + alpha*input          (f32 out)
  gemm_wmma_bf16<1><<<gemm_grid, 256, 0, stream>>>(
      adj, scaledT, support4, nullptr, input, alpha, N, N);
  // 5) supportT = (support4 @ weight)^T               (bf16 out, K-major)
  gemm_wmma_bf16<0><<<gemm_grid, 256, 0, stream>>>(
      support4, Wt, nullptr, supportT, nullptr, nullptr, N, F);
  // 6) out = incidence @ support                      (f32 out)
  gemm_wmma_bf16<2><<<gemm_grid, 256, 0, stream>>>(
      incidence, supportT, out, nullptr, nullptr, nullptr, N, N);
}
